// GraphSageNet_sampler_35493609734997
// MI455X (gfx1250) — matrix operand order repaired and checked by reading, not compiled, not run on hardware
//
#include <hip/hip_runtime.h>
#include <hip/hip_bf16.h>

typedef __attribute__((ext_vector_type(16))) __bf16 bf16x16;
typedef __attribute__((ext_vector_type(8)))  __bf16 bf16x8;
typedef __attribute__((ext_vector_type(8)))  float  f32x8;
typedef __attribute__((ext_vector_type(4)))  float  f32x4;
typedef __attribute__((ext_vector_type(4)))  unsigned int u32x4;
typedef __attribute__((ext_vector_type(8)))  int i32x8;
typedef __attribute__((ext_vector_type(4)))  int i32x4;

#define HID 256
#define MAX_SAGE_BLOCKS 2048   // layer 0: 65536 rows / 32 rows per block

#if defined(__has_builtin)
#if __has_builtin(__builtin_amdgcn_tensor_load_to_lds) && \
    __has_builtin(__builtin_amdgcn_s_wait_tensorcnt)
#define HAVE_TDM 1
#endif
#endif

__device__ __forceinline__ bf16x16 bf_zero() {
  bf16x16 v;
#pragma unroll
  for (int i = 0; i < 16; ++i) v[i] = (__bf16)0.0f;
  return v;
}

__device__ __forceinline__ f32x8 f_zero() {
  f32x8 v;
#pragma unroll
  for (int i = 0; i < 8; ++i) v[i] = 0.0f;
  return v;
}

// A-tile (16x32 bf16) from a row-major [*, HID] panel at (rowBase, kstep).
// A layout (contiguous K, matches load_b_tile): lanes 0-15 row=lane, K=+0..15;
// lanes 16-31 row=lane-16, K=+16..31.
__device__ __forceinline__ bf16x16 load_a_tile(const __bf16* M, int rowBase, int kstep) {
  const int lane = threadIdx.x & 31;
  const int r  = rowBase + (lane & 15);
  const int kb = (lane >> 4) << 4;              // 0 or 16
  const __bf16* p = M + (size_t)r * HID + kstep + kb;
  bf16x8 lo = *(const bf16x8*)(p);
  bf16x8 hi = *(const bf16x8*)(p + 8);
  bf16x16 out;
#pragma unroll
  for (int i = 0; i < 8; ++i) { out[i] = lo[i]; out[8 + i] = hi[i]; }
  return out;
}

// B-tile (32x16 bf16) from K-transposed weights WT[N][K] at (colBase, kstep).
// Lanes 0-15: col=lane, K=kstep+0..15; lanes 16-31: col=lane-16, K=kstep+16..31.
__device__ __forceinline__ bf16x16 load_b_tile(const __bf16* WT, int colBase, int kstep) {
  const int lane = threadIdx.x & 31;
  const int n  = colBase + (lane & 15);
  const int kb = (lane >> 4) << 4;              // 0 or 16
  return *(const bf16x16*)(WT + (size_t)n * HID + kstep + kb);
}

#ifdef HAVE_TDM
__device__ __forceinline__ unsigned int rfl(unsigned int x) {
  return (unsigned int)__builtin_amdgcn_readfirstlane((int)x);
}

// TDM: async 2D bf16 tile (rows x elems, row stride strideElems) global -> LDS.
// D# packing per CDNA5 ISA 8.3/8.4 (group0: count|lds|global|type=2;
// group1: data_size=1(2B), tensor dims, tile dims, dim0 stride).
__device__ __forceinline__ void tdm_load_2d_bf16(unsigned int lds_addr, const void* gptr,
                                                 unsigned int elems, unsigned int rows,
                                                 unsigned int strideElems) {
  const unsigned long long ga = (unsigned long long)(uintptr_t)gptr;
  u32x4 g0;
  g0[0] = rfl(1u);                                                       // count=1 (valid)
  g0[1] = rfl(lds_addr);                                                 // lds_addr
  g0[2] = rfl((unsigned int)ga);                                         // global_addr[31:0]
  g0[3] = rfl(((unsigned int)(ga >> 32) & 0x01FFFFFFu) | (2u << 30));    // [56:32] | type=2
  i32x8 g1;
  g1[0] = (int)rfl(1u << 16);                                            // data_size=2B, mask=0
  g1[1] = (int)rfl((elems & 0xFFFFu) << 16);                             // tensor_dim0 lo
  g1[2] = (int)rfl(((elems >> 16) & 0xFFFFu) | ((rows & 0xFFFFu) << 16));// dim0 hi | dim1 lo
  g1[3] = (int)rfl(((rows >> 16) & 0xFFFFu) | ((elems & 0xFFFFu) << 16));// dim1 hi | tile_dim0
  g1[4] = (int)rfl(rows & 0xFFFFu);                                      // tile_dim1 | tile_dim2=0
  g1[5] = (int)rfl(strideElems);                                         // tensor_dim0_stride
  g1[6] = 0;
  g1[7] = 0;
  i32x4 z4 = {0, 0, 0, 0};
#if __has_include(<hip/amd_detail/amd_gfx1250_TDM.h>)
  i32x8 z8 = {0, 0, 0, 0, 0, 0, 0, 0};
  __builtin_amdgcn_tensor_load_to_lds(g0, g1, z4, z4, z8, 0);
#else
  __builtin_amdgcn_tensor_load_to_lds(g0, g1, z4, z4, 0);
#endif
}
#endif  // HAVE_TDM

// ---------------------------------------------------------------------------
// Weight preparation: f32 row-major [K][N] -> bf16 transposed [N][K]
// ---------------------------------------------------------------------------
__global__ void __launch_bounds__(256) convert_weights_kernel(
    const float* __restrict__ Ws, const float* __restrict__ Wn,
    const float* __restrict__ Wemb,
    __bf16* __restrict__ WsT, __bf16* __restrict__ WnT, __bf16* __restrict__ WembT) {
  const int idx = blockIdx.x * blockDim.x + threadIdx.x;
  if (idx < 3 * 256 * 256) {
    const int l = idx >> 16;
    const int n = (idx >> 8) & 255;
    const int k = idx & 255;
    WsT[idx] = (__bf16)Ws[(size_t)l * 65536 + k * 256 + n];
    WnT[idx] = (__bf16)Wn[(size_t)l * 65536 + k * 256 + n];
  }
  if (idx < 256 * 16) {
    const int n = idx >> 4;
    const int k = idx & 15;
    WembT[idx] = (__bf16)Wemb[k * 256 + n];
  }
}

// ---------------------------------------------------------------------------
// Embedding: h1[n,256] = h0[n,16] @ W_emb + b_emb  (K=16 zero-padded to 32).
// No B masking needed: the A-tile's K=16..31 half is zero, and the values B
// carries in lanes >=16 (K=16..31) are finite weight data, so 0*finite = 0.
// Results staged in LDS, then written back with coalesced b128 stores.
// ---------------------------------------------------------------------------
__global__ void __launch_bounds__(256) embed_kernel(
    const float* __restrict__ h0, const __bf16* __restrict__ WembT,
    const float* __restrict__ b_emb, __bf16* __restrict__ h1) {
  __shared__ __bf16 hs[128 * HID];                 // 64 KB staging tile
  const int w    = threadIdx.x >> 5;
  const int lane = threadIdx.x & 31;
  const int rowBase = blockIdx.x * 128 + w * 16;
  const int r  = rowBase + (lane & 15);
  const int kb = (lane >> 4) << 3;                 // 0 or 8 (< 16 -> all real K)
  bf16x16 a = bf_zero();
  const float* hp = h0 + (size_t)r * 16 + kb;
#pragma unroll
  for (int i = 0; i < 16; ++i) if (lane < 16) a[i] = (__bf16)hp[i];  // lanes>=16 (K=16..31) stay zero
  const int col   = lane & 15;
  const int r0Loc = w * 16 + ((lane >> 4) << 3);
#pragma unroll
  for (int c = 0; c < 16; ++c) {
    // All lanes load in-bounds rows; lanes >=16 contribute only against A's
    // zero half, so no masking/select is required.
    bf16x16 b = *(const bf16x16*)(WembT + (size_t)(c * 16 + col) * 16);
    f32x8 acc = f_zero();
    acc = __builtin_amdgcn_wmma_f32_16x16x32_bf16(false, a, false, b, (short)0, acc,
                                                  false, false);
    const int cc = c * 16 + col;
    const float bias = b_emb[cc];
#pragma unroll
    for (int i = 0; i < 8; ++i)
      hs[(r0Loc + i) * HID + cc] = (__bf16)(acc[i] + bias);
  }
  __syncthreads();
  const u32x4* s4 = (const u32x4*)hs;
  u32x4* d4 = (u32x4*)(h1 + (size_t)blockIdx.x * 128 * HID);
#pragma unroll
  for (int j = 0; j < 16; ++j)
    d4[j * 256 + threadIdx.x] = s4[j * 256 + threadIdx.x];
}

// ---------------------------------------------------------------------------
// Neighbor mean: dst d averages 8 contiguous src entries (dst=repeat(arange,8)).
// Block-wide column access of each gathered row is fully coalesced (512B/row).
// ---------------------------------------------------------------------------
__global__ void __launch_bounds__(256) mean_kernel(
    const __bf16* __restrict__ h, const int* __restrict__ src,
    __bf16* __restrict__ meanb, int n_dst) {
  const int d = blockIdx.x;
  const int t = threadIdx.x;          // column 0..255
  const int* sp = src + (size_t)d * 8;
  float s = 0.f;
#pragma unroll
  for (int k = 0; k < 8; ++k) s += (float)h[(size_t)sp[k] * HID + t];
  meanb[(size_t)d * HID + t] = (__bf16)(s * 0.125f);
  (void)n_dst;
}

// ---------------------------------------------------------------------------
// Z[n,256] = Hdst @ Ws + Mean @ Wn + b, fused with per-block BN partials.
// A-panels (32 rows of hdst + mean) staged in LDS via TDM (wave 0 issues
// tensor_load_to_lds, waits TENSORcnt, then workgroup barrier); B from global
// (bf16 weights are L2-resident). 8 WMMAs per wave per K-step. Output tile
// staged in LDS (reusing A staging space), stored coalesced; per-column
// sum/sum^2 partials are accumulated during the copy (each thread's 4 columns
// are fixed since all strides are multiples of 256) -> no extra Z read pass.
// ---------------------------------------------------------------------------
__global__ void __launch_bounds__(256) sage_wmma_kernel(
    const __bf16* __restrict__ hdst, const __bf16* __restrict__ meanb,
    const __bf16* __restrict__ WsT, const __bf16* __restrict__ WnT,
    const float* __restrict__ bias, float* __restrict__ Z,
    float* __restrict__ partials, int n) {
  __shared__ float smem[32 * HID];                 // 32 KB: A staging, then f32 out
  __shared__ float ps[4 * 256];                    // per-group column sums
  __shared__ float pss[4 * 256];                   // per-group column sum-squares
  __bf16* tileH = (__bf16*)smem;                   // [32][256] bf16 (16 KB)
  __bf16* tileM = tileH + 32 * HID;                // [32][256] bf16 (16 KB)
  const int rowBlock = blockIdx.x * 32;
#ifdef HAVE_TDM
  if (threadIdx.x < 32) {
    tdm_load_2d_bf16((unsigned int)(uintptr_t)tileH,
                     hdst + (size_t)rowBlock * HID, HID, 32, HID);
    tdm_load_2d_bf16((unsigned int)(uintptr_t)tileM,
                     meanb + (size_t)rowBlock * HID, HID, 32, HID);
    __builtin_amdgcn_s_wait_tensorcnt(0);
  }
#else
  {
    const unsigned int* gh = (const unsigned int*)(hdst + (size_t)rowBlock * HID);
    const unsigned int* gm = (const unsigned int*)(meanb + (size_t)rowBlock * HID);
    unsigned int* lh = (unsigned int*)tileH;
    unsigned int* lm = (unsigned int*)tileM;
#pragma unroll
    for (int j = 0; j < 16; ++j) {
      lh[j * 256 + threadIdx.x] = gh[j * 256 + threadIdx.x];
      lm[j * 256 + threadIdx.x] = gm[j * 256 + threadIdx.x];
    }
  }
#endif
  __syncthreads();

  const int w = threadIdx.x >> 5;
  const int rLoc    = (w & 1) << 4;                // 0 or 16 within the 32-row panel
  const int colBase = (w >> 1) << 6;               // 0/64/128/192
  f32x8 acc[4];
#pragma unroll
  for (int c = 0; c < 4; ++c) acc[c] = f_zero();
#pragma unroll
  for (int kstep = 0; kstep < HID; kstep += 32) {
    bf16x16 ah = load_a_tile(tileH, rLoc, kstep);  // ds_load_b128 x2
    bf16x16 am = load_a_tile(tileM, rLoc, kstep);
#pragma unroll
    for (int c = 0; c < 4; ++c) {
      bf16x16 bs = load_b_tile(WsT, colBase + c * 16, kstep);
      acc[c] = __builtin_amdgcn_wmma_f32_16x16x32_bf16(false, ah, false, bs, (short)0,
                                                       acc[c], false, false);
      bf16x16 bn = load_b_tile(WnT, colBase + c * 16, kstep);
      acc[c] = __builtin_amdgcn_wmma_f32_16x16x32_bf16(false, am, false, bn, (short)0,
                                                       acc[c], false, false);
    }
  }
  __syncthreads();                                 // A panels consumed; reuse smem

  const int lane = threadIdx.x & 31;
  const int col0 = colBase + (lane & 15);
  const int rOff = rLoc + ((lane >> 4) << 3);
#pragma unroll
  for (int c = 0; c < 4; ++c) {
    const int cc = col0 + c * 16;
    const float bv = bias[cc];
#pragma unroll
    for (int i = 0; i < 8; ++i)
      smem[(rOff + i) * HID + cc] = acc[c][i] + bv;
  }
  __syncthreads();

  // Coalesced store + fused BN partial accumulation.
  const int t = threadIdx.x;
  const f32x4* s4 = (const f32x4*)smem;
  f32x4* d4 = (f32x4*)(Z + (size_t)rowBlock * HID);
  float s[4] = {0.f, 0.f, 0.f, 0.f};
  float ss[4] = {0.f, 0.f, 0.f, 0.f};
#pragma unroll
  for (int j = 0; j < 8; ++j) {
    const f32x4 v = s4[j * 256 + t];
    d4[j * 256 + t] = v;
#pragma unroll
    for (int e = 0; e < 4; ++e) { s[e] += v[e]; ss[e] += v[e] * v[e]; }
  }
  const int g = t >> 6;                            // thread-group 0..3 shares columns
#pragma unroll
  for (int e = 0; e < 4; ++e) {
    const int cc = (t * 4 + e) & 255;
    ps[g * 256 + cc]  = s[e];
    pss[g * 256 + cc] = ss[e];
  }
  __syncthreads();
  float S  = ps[t] + ps[256 + t] + ps[512 + t] + ps[768 + t];
  float SS = pss[t] + pss[256 + t] + pss[512 + t] + pss[768 + t];
  partials[(size_t)blockIdx.x * 512 + t]       = S;
  partials[(size_t)blockIdx.x * 512 + 256 + t] = SS;
  (void)n;
}

// ---------------------------------------------------------------------------
// BN stats finalize: fixed-order reduction of per-block partials (bitwise
// deterministic, no float atomics) -> per-column (scale, shift).
// ---------------------------------------------------------------------------
__global__ void __launch_bounds__(256) finalize_stats_kernel(
    const float* __restrict__ partials, int nblocks, int n,
    const float* __restrict__ gamma, const float* __restrict__ beta,
    float2* __restrict__ stats) {
  const int col = threadIdx.x;
  float s = 0.f, ss = 0.f;
  for (int b = 0; b < nblocks; ++b) {
    s  += partials[(size_t)b * 512 + col];
    ss += partials[(size_t)b * 512 + 256 + col];
  }
  const float inv_n = 1.0f / (float)n;
  const float mu  = s * inv_n;
  const float var = ss * inv_n - mu * mu;      // biased, matches torch BN training
  const float sc  = gamma[col] * rsqrtf(var + 1e-5f);
  stats[col] = make_float2(sc, beta[col] - mu * sc);
}

__global__ void __launch_bounds__(256) bn_apply_kernel(
    const float* __restrict__ Z, const float2* __restrict__ stats,
    __bf16* __restrict__ hout, int n, int doRelu) {
  const int i = blockIdx.x * blockDim.x + threadIdx.x;
  if (i >= n * HID) return;
  const float2 st = stats[i & 255];
  float v = Z[i] * st.x + st.y;
  if (doRelu) v = fmaxf(v, 0.f);
  hout[i] = (__bf16)v;
}

// ---------------------------------------------------------------------------
// Small dense head (n=1024, tiny FLOPs -> plain VALU)
// ---------------------------------------------------------------------------
__global__ void __launch_bounds__(256) dense_bf16_kernel(
    const __bf16* __restrict__ X, const float* __restrict__ W,
    const float* __restrict__ b, float* __restrict__ Y,
    int n, int K, int N, int doRelu) {
  const int i = blockIdx.x * blockDim.x + threadIdx.x;
  if (i >= n * N) return;
  const int r = i / N, c = i - r * N;
  float s = b[c];
  for (int k = 0; k < K; ++k) s += (float)X[(size_t)r * K + k] * W[(size_t)k * N + c];
  Y[i] = doRelu ? fmaxf(s, 0.f) : s;
}

__global__ void __launch_bounds__(256) dense_f32_kernel(
    const float* __restrict__ X, const float* __restrict__ W,
    const float* __restrict__ b, float* __restrict__ Y,
    int n, int K, int N, int doRelu) {
  const int i = blockIdx.x * blockDim.x + threadIdx.x;
  if (i >= n * N) return;
  const int r = i / N, c = i - r * N;
  float s = b[c];
  for (int k = 0; k < K; ++k) s += X[(size_t)r * K + k] * W[(size_t)k * N + c];
  Y[i] = doRelu ? fmaxf(s, 0.f) : s;
}

// ---------------------------------------------------------------------------
extern "C" void kernel_launch(void* const* d_in, const int* in_sizes, int n_in,
                              void* d_out, int out_size, void* d_ws, size_t ws_size,
                              hipStream_t stream) {
  const float* h0       = (const float*)d_in[0];
  const int*   src0     = (const int*)d_in[1];
  const int*   src1     = (const int*)d_in[3];
  const int*   src2     = (const int*)d_in[5];
  const float* W_emb    = (const float*)d_in[7];
  const float* b_emb    = (const float*)d_in[8];
  const float* W_self   = (const float*)d_in[9];
  const float* W_neigh  = (const float*)d_in[10];
  const float* b_conv   = (const float*)d_in[11];
  const float* bn_gamma = (const float*)d_in[12];
  const float* bn_beta  = (const float*)d_in[13];
  const float* mlp_W0   = (const float*)d_in[14];
  const float* mlp_b0   = (const float*)d_in[15];
  const float* mlp_W1   = (const float*)d_in[16];
  const float* mlp_b1   = (const float*)d_in[17];
  const float* mlp_W2   = (const float*)d_in[18];
  const float* mlp_b2   = (const float*)d_in[19];
  float* out = (float*)d_out;
  (void)in_sizes; (void)n_in; (void)out_size; (void)ws_size;

  char* ws = (char*)d_ws;
  size_t off = 0;
  auto carve = [&](size_t bytes) {
    char* p = ws + off;
    off = (off + bytes + 255) & ~(size_t)255;
    return p;
  };
  __bf16* WembT = (__bf16*)carve((size_t)256 * 16 * 2);
  __bf16* WsT   = (__bf16*)carve((size_t)3 * 256 * 256 * 2);
  __bf16* WnT   = (__bf16*)carve((size_t)3 * 256 * 256 * 2);
  __bf16* h1    = (__bf16*)carve((size_t)524288 * 256 * 2);
  __bf16* h2    = (__bf16*)carve((size_t)65536 * 256 * 2);
  __bf16* h3    = (__bf16*)carve((size_t)8192 * 256 * 2);
  __bf16* h4    = (__bf16*)carve((size_t)1024 * 256 * 2);
  __bf16* meanb = (__bf16*)carve((size_t)65536 * 256 * 2);
  float*  Z     = (float*)carve((size_t)65536 * 256 * 4);
  float*  parts = (float*)carve((size_t)MAX_SAGE_BLOCKS * 512 * 4);
  float2* stats = (float2*)carve((size_t)256 * sizeof(float2));
  float*  y0    = (float*)carve((size_t)1024 * 128 * 4);
  float*  y1    = (float*)carve((size_t)1024 * 64 * 4);

  convert_weights_kernel<<<768, 256, 0, stream>>>(W_self, W_neigh, W_emb, WsT, WnT, WembT);

  embed_kernel<<<524288 / 128, 256, 0, stream>>>(h0, WembT, b_emb, h1);

  // layer 0: 524288 -> 65536
  mean_kernel<<<65536, 256, 0, stream>>>(h1, src0, meanb, 65536);
  sage_wmma_kernel<<<65536 / 32, 256, 0, stream>>>(h1, meanb, WsT, WnT, b_conv, Z,
                                                   parts, 65536);
  finalize_stats_kernel<<<1, 256, 0, stream>>>(parts, 65536 / 32, 65536, bn_gamma,
                                               bn_beta, stats);
  bn_apply_kernel<<<(65536 * 256) / 256, 256, 0, stream>>>(Z, stats, h2, 65536, 1);

  // layer 1: 65536 -> 8192
  mean_kernel<<<8192, 256, 0, stream>>>(h2, src1, meanb, 8192);
  sage_wmma_kernel<<<8192 / 32, 256, 0, stream>>>(h2, meanb, WsT + 65536, WnT + 65536,
                                                  b_conv + 256, Z, parts, 8192);
  finalize_stats_kernel<<<1, 256, 0, stream>>>(parts, 8192 / 32, 8192, bn_gamma + 256,
                                               bn_beta + 256, stats);
  bn_apply_kernel<<<(8192 * 256) / 256, 256, 0, stream>>>(Z, stats, h3, 8192, 1);

  // layer 2: 8192 -> 1024 (no ReLU after BN)
  mean_kernel<<<1024, 256, 0, stream>>>(h3, src2, meanb, 1024);
  sage_wmma_kernel<<<1024 / 32, 256, 0, stream>>>(h3, meanb, WsT + 2 * 65536,
                                                  WnT + 2 * 65536, b_conv + 512, Z,
                                                  parts, 1024);
  finalize_stats_kernel<<<1, 256, 0, stream>>>(parts, 1024 / 32, 1024, bn_gamma + 512,
                                               bn_beta + 512, stats);
  bn_apply_kernel<<<(1024 * 256) / 256, 256, 0, stream>>>(Z, stats, h4, 1024, 0);

  // MLP head
  dense_bf16_kernel<<<(1024 * 128) / 256, 256, 0, stream>>>(h4, mlp_W0, mlp_b0, y0,
                                                            1024, 256, 128, 1);
  dense_f32_kernel<<<(1024 * 64) / 256, 256, 0, stream>>>(y0, mlp_W1, mlp_b1, y1,
                                                          1024, 128, 64, 1);
  dense_f32_kernel<<<(1024 * 21 + 255) / 256, 256, 0, stream>>>(y1, mlp_W2, mlp_b2, out,
                                                                1024, 64, 21, 0);
}
